// LSTMDecoder_2645699855076
// MI455X (gfx1250) — matrix operand order repaired and checked by reading, not compiled, not run on hardware
//
#include <hip/hip_runtime.h>
#include <hip/hip_bf16.h>

// Problem constants (reference: B,M,E,H,S,F = 2048,128,256,256,64,4)
#define BB 2048
#define MM 128
#define EE 256
#define HH 256
#define SS 64
#define FF 4

typedef __attribute__((ext_vector_type(16))) __bf16 v16bf;
typedef __attribute__((ext_vector_type(8)))  __bf16 bf16x8;
typedef __attribute__((ext_vector_type(8)))  float  v8f;

union Frag16 { v16bf v; bf16x8 h[2]; };

__device__ __forceinline__ float sigf(float x) { return 1.f / (1.f + __expf(-x)); }

// ---------------------------------------------------------------------------
// Pack weights to bf16 (K-contiguous, matching WMMA B-fragment loads) + biases
// ---------------------------------------------------------------------------
__global__ void k_prep(const float* __restrict__ W_ih0, const float* __restrict__ W_hh0,
                       const float* __restrict__ b_ih0, const float* __restrict__ b_hh0,
                       const float* __restrict__ W_ih1, const float* __restrict__ W_hh1,
                       const float* __restrict__ b_ih1, const float* __restrict__ b_hh1,
                       const float* __restrict__ W_att,
                       __bf16* __restrict__ Wcat0, __bf16* __restrict__ Wcat1,
                       __bf16* __restrict__ Wattb,
                       float* __restrict__ bsum0, float* __restrict__ bsum1) {
  int idx = blockIdx.x * blockDim.x + threadIdx.x;
  const int WN = 4 * HH * (EE + HH); // 1024*512
  if (idx < WN) {
    int n = idx >> 9, k = idx & 511;
    float v = (k < EE) ? W_ih0[n * EE + k] : W_hh0[n * HH + (k - EE)];
    Wcat0[idx] = (__bf16)v;
    return;
  }
  idx -= WN;
  if (idx < WN) {
    int n = idx >> 9, k = idx & 511;
    float v = (k < HH) ? W_ih1[n * HH + k] : W_hh1[n * HH + (k - HH)];
    Wcat1[idx] = (__bf16)v;
    return;
  }
  idx -= WN;
  if (idx < HH * (EE + HH)) { // W_att is already [H, E+H] = [o1|ctx] order
    Wattb[idx] = (__bf16)W_att[idx];
    return;
  }
  idx -= HH * (EE + HH);
  if (idx < 4 * HH) { bsum0[idx] = b_ih0[idx] + b_hh0[idx]; return; }
  idx -= 4 * HH;
  if (idx < 4 * HH) { bsum1[idx] = b_ih1[idx] + b_hh1[idx]; return; }
}

// ---------------------------------------------------------------------------
// A0 = [ memory[b, read_ind[b]] | h0 ]  (bf16, [B, 512])
// ---------------------------------------------------------------------------
__global__ void k_gather(const float* __restrict__ memory, const float* __restrict__ h0,
                         const int* __restrict__ read_ind, __bf16* __restrict__ A0) {
  int idx = blockIdx.x * blockDim.x + threadIdx.x; // B*512
  int b = idx >> 9, k = idx & 511;
  float v = (k < EE) ? memory[((size_t)b * MM + read_ind[b]) * EE + k]
                     : h0[(size_t)b * HH + (k - EE)];
  A0[idx] = (__bf16)v;
}

// ---------------------------------------------------------------------------
// WMMA bf16 GEMM: C[Mr,N] = A[Mr,512] * W[N,512]^T (+bias) (optional tanh)
// One wave computes a 32x64 tile (8 accumulators). K loop fully unrolled with
// explicit ping-pong fragment buffers so loads for step k+32 are in flight
// while the 8 WMMAs of step k execute; no rotation copies, no spills
// (__launch_bounds__(128,1) gives the wave its full VGPR budget).
// Fragment layouts per CDNA5 ISA 7.12.2.
// ---------------------------------------------------------------------------
__global__ void __launch_bounds__(128, 1)
k_gemm(const __bf16* __restrict__ A, const __bf16* __restrict__ W,
       const float* __restrict__ bias, float* __restrict__ C,
       int Mr, int N, int fuse_tanh) {
  constexpr int K = 512;
  const int lane = threadIdx.x & 31;
  const int wid  = (blockIdx.x * blockDim.x + threadIdx.x) >> 5;
  const int nGroups = N >> 6;               // 64-column groups
  const int tm = wid / nGroups;             // 32-row tile index
  const int tg = wid % nGroups;             // 64-col group index
  if (tm * 32 >= Mr) return;

  // A fragments: rows tm*32 + lane%16 and +16.
  // lanes<16 hold K {0..7,16..23}, lanes>=16 hold K {8..15,24..31}
  const int kloA = (lane & 16) ? 8 : 0;
  const __bf16* aRow0 = A + (size_t)(tm * 32 + (lane & 15)) * K + kloA;
  const __bf16* aRow1 = aRow0 + (size_t)16 * K;

  // B fragments: col = tg*64 + j*16 + lane%16; lanes<16 K {0..7,16..23}, lanes>=16 K {8..15,24..31}
  const int kloB = (lane & 16) ? 8 : 0;
  const __bf16* bRow[4];
#pragma unroll
  for (int j = 0; j < 4; ++j)
    bRow[j] = W + (size_t)(tg * 64 + j * 16 + (lane & 15)) * K + kloB;

  v8f acc[8] = {};
  Frag16 aP[2], bP[4];   // ping
  Frag16 aQ[2], bQ[4];   // pong

#define LOAD_FRAGS(AF, BF, KOFF)                                               \
  do {                                                                         \
    (AF)[0].h[0] = *reinterpret_cast<const bf16x8*>(aRow0 + (KOFF));           \
    (AF)[0].h[1] = *reinterpret_cast<const bf16x8*>(aRow0 + (KOFF) + 16);      \
    (AF)[1].h[0] = *reinterpret_cast<const bf16x8*>(aRow1 + (KOFF));           \
    (AF)[1].h[1] = *reinterpret_cast<const bf16x8*>(aRow1 + (KOFF) + 16);      \
    _Pragma("unroll") for (int j = 0; j < 4; ++j) {                            \
      (BF)[j].h[0] = *reinterpret_cast<const bf16x8*>(bRow[j] + (KOFF));       \
      (BF)[j].h[1] = *reinterpret_cast<const bf16x8*>(bRow[j] + (KOFF) + 16);  \
    }                                                                          \
  } while (0)

#define DO_WMMA(AF, BF)                                                        \
  do {                                                                         \
    _Pragma("unroll") for (int j = 0; j < 4; ++j) {                            \
      acc[j] = __builtin_amdgcn_wmma_f32_16x16x32_bf16(                        \
          false, (AF)[0].v, false, (BF)[j].v, (short)0, acc[j], false, false); \
      acc[4 + j] = __builtin_amdgcn_wmma_f32_16x16x32_bf16(                    \
          false, (AF)[1].v, false, (BF)[j].v, (short)0, acc[4 + j], false,     \
          false);                                                              \
    }                                                                          \
  } while (0)

  LOAD_FRAGS(aP, bP, 0);
#pragma unroll
  for (int k0 = 0; k0 < K; k0 += 64) {
    LOAD_FRAGS(aQ, bQ, k0 + 32);        // in flight while ping WMMAs run
    DO_WMMA(aP, bP);
    if (k0 + 64 < K) LOAD_FRAGS(aP, bP, k0 + 64);
    DO_WMMA(aQ, bQ);
  }
#undef LOAD_FRAGS
#undef DO_WMMA

  // C/D layout: VGPR r -> M = r (lanes 0-15) / r+8 (lanes 16-31); N = lane%16
  const int rbase = (lane & 16) ? 8 : 0;
  const int coll  = lane & 15;
#pragma unroll
  for (int j = 0; j < 4; ++j) {
    const int col = tg * 64 + j * 16 + coll;
    const float bv = bias ? bias[col] : 0.f;
#pragma unroll
    for (int r = 0; r < 8; ++r) {
      float v0 = acc[j][r] + bv;
      float v1 = acc[4 + j][r] + bv;
      if (fuse_tanh) { v0 = tanhf(v0); v1 = tanhf(v1); }
      C[(size_t)(tm * 32 + rbase + r) * N + col]      = v0;
      C[(size_t)(tm * 32 + 16 + rbase + r) * N + col] = v1;
    }
  }
}

// ---------------------------------------------------------------------------
// LSTM pointwise layer 0: gates0 -> h0n,c0n; build A1 = [h0n | h1] bf16
// ---------------------------------------------------------------------------
__global__ void k_lstm0(const float* __restrict__ gates, const float* __restrict__ c_in,
                        const float* __restrict__ h1_in,
                        float* __restrict__ h_out, float* __restrict__ c_out,
                        __bf16* __restrict__ A1) {
  int idx = blockIdx.x * blockDim.x + threadIdx.x; // B*H
  int b = idx >> 8, h = idx & 255;
  const float* g = gates + (size_t)b * (4 * HH);
  float gi = sigf(g[h]);
  float gf = sigf(g[HH + h]);
  float gg = tanhf(g[2 * HH + h]);
  float go = sigf(g[3 * HH + h]);
  float c_new = gf * c_in[idx] + gi * gg;
  float h_new = go * tanhf(c_new);
  h_out[idx] = h_new;
  c_out[idx] = c_new;
  A1[(size_t)b * 512 + h]       = (__bf16)h_new;
  A1[(size_t)b * 512 + HH + h]  = (__bf16)h1_in[idx];
}

// ---------------------------------------------------------------------------
// LSTM pointwise layer 1: gates1 -> h1n,c1n; o1 = h1n + h0n; A2[:,0:256]=o1
// ---------------------------------------------------------------------------
__global__ void k_lstm1(const float* __restrict__ gates, const float* __restrict__ c_in,
                        const float* __restrict__ h0n,
                        float* __restrict__ h_out, float* __restrict__ c_out,
                        float* __restrict__ o1, __bf16* __restrict__ A2) {
  int idx = blockIdx.x * blockDim.x + threadIdx.x; // B*H
  int b = idx >> 8, h = idx & 255;
  const float* g = gates + (size_t)b * (4 * HH);
  float gi = sigf(g[h]);
  float gf = sigf(g[HH + h]);
  float gg = tanhf(g[2 * HH + h]);
  float go = sigf(g[3 * HH + h]);
  float c_new = gf * c_in[idx] + gi * gg;
  float h_new = go * tanhf(c_new);
  h_out[idx] = h_new;
  c_out[idx] = c_new;
  float o1v = h_new + h0n[idx];
  o1[idx] = o1v;
  A2[(size_t)b * 512 + h] = (__bf16)o1v;
}

// ---------------------------------------------------------------------------
// Attention: att_w = keys . o1 (masked), softmax, ctx = p . values
// One block (256 thr = 8 waves) per batch. wave32 shuffle reductions.
// ---------------------------------------------------------------------------
__global__ void k_attn(const float* __restrict__ keys, const float* __restrict__ values,
                       const float* __restrict__ qmask, const float* __restrict__ o1,
                       __bf16* __restrict__ A2) {
  const int b = blockIdx.x;
  __shared__ float s_o1[HH];
  __shared__ float s_w[SS];
  __shared__ float s_p[SS];
  const int t = threadIdx.x;
  s_o1[t] = o1[(size_t)b * HH + t];
  __syncthreads();

  const int wave = t >> 5, lane = t & 31;
#pragma unroll
  for (int r = 0; r < 8; ++r) {
    const int s = wave * 8 + r;
    const float* kp = keys + ((size_t)b * SS + s) * HH + lane * 8;
    float acc = 0.f;
#pragma unroll
    for (int j = 0; j < 8; ++j) acc += kp[j] * s_o1[lane * 8 + j];
    for (int off = 16; off; off >>= 1) acc += __shfl_xor(acc, off);
    if (lane == 0) {
      float m = qmask[(size_t)b * SS + s];
      s_w[s] = (m > 0.f) ? acc : -__builtin_inff();
    }
  }
  __syncthreads();

  float mx = -__builtin_inff();
  for (int s = 0; s < SS; ++s) mx = fmaxf(mx, s_w[s]);
  if (t < SS) s_p[t] = __expf(s_w[t] - mx);
  __syncthreads();
  float sum = 0.f;
  for (int s = 0; s < SS; ++s) sum += s_p[s];
  const float inv = 1.f / sum;

  // ctx[e], e = t (256 threads)
  float ctx = 0.f;
  const float* vp = values + (size_t)b * SS * EE + t;
  for (int s = 0; s < SS; ++s) ctx += s_p[s] * vp[(size_t)s * EE];
  A2[(size_t)b * 512 + HH + t] = (__bf16)(ctx * inv);
}

// ---------------------------------------------------------------------------
// Fused final pass: read memory once -> mem_logits (dot with att_t, + of.W_of)
// and memory_t (conditional scatter-add of att_t at write_ind).
// One wave per (b, m) row; b128 loads/stores.
// ---------------------------------------------------------------------------
__global__ void k_final(const float* __restrict__ memory, const float* __restrict__ att_t,
                        const float* __restrict__ of, const float* __restrict__ W_of,
                        const int* __restrict__ write_ind,
                        float* __restrict__ mem_logits, float* __restrict__ memory_t) {
  const int gw   = (blockIdx.x * blockDim.x + threadIdx.x) >> 5; // row id
  const int lane = threadIdx.x & 31;
  const int b = gw >> 7;        // / M
  const int m = gw & (MM - 1);

  const size_t rowOff = ((size_t)b * MM + m) * EE;
  const float4* mp = reinterpret_cast<const float4*>(memory + rowOff + lane * 8);
  const float4* at = reinterpret_cast<const float4*>(att_t + (size_t)b * HH + lane * 8);

  const int wiRaw = write_ind[b];
  const float wm  = (wiRaw >= 0) ? 1.f : 0.f;
  const int wrow  = (wiRaw >= 0) ? wiRaw : 0;
  const float addv = (m == wrow) ? wm : 0.f;

  float4 m0 = mp[0], m1 = mp[1];
  float4 a0 = at[0], a1 = at[1];
  float dot = m0.x * a0.x + m0.y * a0.y + m0.z * a0.z + m0.w * a0.w
            + m1.x * a1.x + m1.y * a1.y + m1.z * a1.z + m1.w * a1.w;
  float4 o0 = make_float4(m0.x + addv * a0.x, m0.y + addv * a0.y,
                          m0.z + addv * a0.z, m0.w + addv * a0.w);
  float4 o1v = make_float4(m1.x + addv * a1.x, m1.y + addv * a1.y,
                           m1.z + addv * a1.z, m1.w + addv * a1.w);
  float4* op = reinterpret_cast<float4*>(memory_t + rowOff + lane * 8);
  op[0] = o0; op[1] = o1v;

  for (int off = 16; off; off >>= 1) dot += __shfl_xor(dot, off);
  if (lane == 0) {
    const float* fp = of + ((size_t)b * MM + m) * FF;
    float sc = fp[0] * W_of[0] + fp[1] * W_of[1] + fp[2] * W_of[2] + fp[3] * W_of[3];
    mem_logits[(size_t)b * MM + m] = dot + sc;
  }
}

// ---------------------------------------------------------------------------
extern "C" void kernel_launch(void* const* d_in, const int* in_sizes, int n_in,
                              void* d_out, int out_size, void* d_ws, size_t ws_size,
                              hipStream_t stream) {
  const float* memory = (const float*)d_in[0];
  const float* h0     = (const float*)d_in[1];
  const float* c0     = (const float*)d_in[2];
  const float* h1     = (const float*)d_in[3];
  const float* c1     = (const float*)d_in[4];
  const float* keys   = (const float*)d_in[5];
  const float* values = (const float*)d_in[6];
  const float* qmask  = (const float*)d_in[7];
  const float* of     = (const float*)d_in[8];
  const float* W_ih0  = (const float*)d_in[9];
  const float* W_hh0  = (const float*)d_in[10];
  const float* b_ih0  = (const float*)d_in[11];
  const float* b_hh0  = (const float*)d_in[12];
  const float* W_ih1  = (const float*)d_in[13];
  const float* W_hh1  = (const float*)d_in[14];
  const float* b_ih1  = (const float*)d_in[15];
  const float* b_hh1  = (const float*)d_in[16];
  const float* W_att  = (const float*)d_in[17];
  const float* W_of   = (const float*)d_in[18];
  const int* read_ind  = (const int*)d_in[19];
  const int* write_ind = (const int*)d_in[20];

  // Output layout: mem_logits | memory_t | h0n | c0n | h1n | c1n
  float* out        = (float*)d_out;
  float* mem_logits = out;
  float* memory_t   = out + (size_t)BB * MM;
  float* h0n        = memory_t + (size_t)BB * MM * EE;
  float* c0n        = h0n + (size_t)BB * HH;
  float* h1n        = c0n + (size_t)BB * HH;
  float* c1n        = h1n + (size_t)BB * HH;

  // Workspace layout (all chunks 256B-aligned)
  char* ws = (char*)d_ws;
  size_t off = 0;
  __bf16* Wcat0 = (__bf16*)(ws + off); off += (size_t)4 * HH * 512 * 2;
  __bf16* Wcat1 = (__bf16*)(ws + off); off += (size_t)4 * HH * 512 * 2;
  __bf16* Wattb = (__bf16*)(ws + off); off += (size_t)HH * 512 * 2;
  float*  bsum0 = (float*)(ws + off);  off += (size_t)4 * HH * 4;
  float*  bsum1 = (float*)(ws + off);  off += (size_t)4 * HH * 4;
  __bf16* A0    = (__bf16*)(ws + off); off += (size_t)BB * 512 * 2;
  __bf16* A1    = (__bf16*)(ws + off); off += (size_t)BB * 512 * 2;
  __bf16* A2    = (__bf16*)(ws + off); off += (size_t)BB * 512 * 2;
  float*  gates0 = (float*)(ws + off); off += (size_t)BB * 4 * HH * 4;
  float*  gates1 = (float*)(ws + off); off += (size_t)BB * 4 * HH * 4;
  float*  o1f    = (float*)(ws + off); off += (size_t)BB * HH * 4;
  float*  att_t  = (float*)(ws + off); off += (size_t)BB * HH * 4;

  // 1) weight packing + bias sums
  {
    int total = 2 * (4 * HH * 512) + HH * 512 + 2 * (4 * HH);
    k_prep<<<(total + 255) / 256, 256, 0, stream>>>(
        W_ih0, W_hh0, b_ih0, b_hh0, W_ih1, W_hh1, b_ih1, b_hh1, W_att,
        Wcat0, Wcat1, Wattb, bsum0, bsum1);
  }
  // 2) gather + pack A0
  k_gather<<<(BB * 512) / 256, 256, 0, stream>>>(memory, h0, read_ind, A0);
  // 3) gates0 = A0 * Wcat0^T + bsum0      (WMMA bf16, 32x64 tile/wave)
  {
    int waves = (BB / 32) * (4 * HH / 64);
    k_gemm<<<waves / 4, 128, 0, stream>>>(A0, Wcat0, bsum0, gates0, BB, 4 * HH, 0);
  }
  // 4) LSTM0 pointwise
  k_lstm0<<<(BB * HH) / 256, 256, 0, stream>>>(gates0, c0, h1, h0n, c0n, A1);
  // 5) gates1 = A1 * Wcat1^T + bsum1      (WMMA bf16)
  {
    int waves = (BB / 32) * (4 * HH / 64);
    k_gemm<<<waves / 4, 128, 0, stream>>>(A1, Wcat1, bsum1, gates1, BB, 4 * HH, 0);
  }
  // 6) LSTM1 pointwise + skip
  k_lstm1<<<(BB * HH) / 256, 256, 0, stream>>>(gates1, c1, h0n, h1n, c1n, o1f, A2);
  // 7) attention (softmax + ctx) -> A2[:,256:512]
  k_attn<<<BB, 256, 0, stream>>>(keys, values, qmask, o1f, A2);
  // 8) att_t = tanh(A2 * Wattb^T)         (WMMA bf16, fused tanh)
  {
    int waves = (BB / 32) * (HH / 64);
    k_gemm<<<waves / 4, 128, 0, stream>>>(A2, Wattb, nullptr, att_t, BB, HH, 1);
  }
  // 9) fused mem_logits + memory_t scatter-write (read memory exactly once)
  {
    int rows = BB * MM;                 // one wave per row
    k_final<<<rows / 8, 256, 0, stream>>>(memory, att_t, of, W_of, write_ind,
                                          mem_logits, memory_t);
  }
}
